// NonLinearDipolePolarReadoutBlock_19164144075358
// MI455X (gfx1250) — matrix operand order repaired and checked by reading, not compiled, not run on hardware
//
#include <hip/hip_runtime.h>
#include <hip/hip_fp16.h>

typedef _Float16 v16h __attribute__((ext_vector_type(16)));
typedef float    v8f  __attribute__((ext_vector_type(8)));
typedef float    v4f  __attribute__((ext_vector_type(4)));

#define D_IN   1152
#define INV_IN 0.08838834764831845f   // 1/sqrt(128)
#define INV_H  0.25f                  // 1/sqrt(16)

__device__ __forceinline__ float silu_f(float v) {
  return v / (1.0f + __expf(-v));
}

__device__ __forceinline__ v8f wmma_f16(v16h a, v16h b, v8f c) {
  // D = A(16x32 f16) * B(32x16 f16) + C(16x16 f32)
  return __builtin_amdgcn_wmma_f32_16x16x32_f16(
      /*neg_a=*/false, a, /*neg_b=*/false, b,
      /*c_mod=*/(short)0, c, /*reuse_a=*/false, /*reuse_b=*/false);
}

// Load one K-run group exactly once (contiguous, 16B-aligned, non-temporal),
// then de-interleave STRIDE components during f32->f16 conversion and issue
// one WMMA per component. A-fragment layout (16-bit A 16x32):
//   lane row m = lane&15, group g = lane>>4, halves 0..7 -> K = u1..u1+7,
//   halves 8..15 -> K = u1+16..u1+23, with u1 = kc*32 + g*8.
// Elements live at row[off + STRIDE*u + comp]; the union over comps of one
// run is the contiguous float range [off+STRIDE*u1, off+STRIDE*(u1+8)).
template <int STRIDE, int NCOMP>
__device__ __forceinline__ void run_group(const float* __restrict__ row,
                                          int off, int u1,
                                          v16h b, v8f* accs) {
  float buf[16 * STRIDE];
  const v4f* p0 = (const v4f*)(row + off + STRIDE * u1);
  const v4f* p1 = (const v4f*)(row + off + STRIDE * (u1 + 16));
#pragma unroll
  for (int t = 0; t < 2 * STRIDE; ++t) {
    v4f v = __builtin_nontemporal_load(p0 + t);
    buf[4 * t + 0] = v[0]; buf[4 * t + 1] = v[1];
    buf[4 * t + 2] = v[2]; buf[4 * t + 3] = v[3];
  }
#pragma unroll
  for (int t = 0; t < 2 * STRIDE; ++t) {
    v4f v = __builtin_nontemporal_load(p1 + t);
    buf[8 * STRIDE + 4 * t + 0] = v[0]; buf[8 * STRIDE + 4 * t + 1] = v[1];
    buf[8 * STRIDE + 4 * t + 2] = v[2]; buf[8 * STRIDE + 4 * t + 3] = v[3];
  }
#pragma unroll
  for (int i = 0; i < NCOMP; ++i) {
    v16h a;
#pragma unroll
    for (int j = 0; j < 8; ++j)
      a[j] = (_Float16)buf[STRIDE * j + i];
#pragma unroll
    for (int j = 0; j < 8; ++j)
      a[j + 8] = (_Float16)buf[8 * STRIDE + STRIDE * j + i];
    accs[i] = wmma_f16(a, b, accs[i]);
  }
}

__global__ __launch_bounds__(256) void readout_wmma_kernel(
    const float* __restrict__ x,
    const float* __restrict__ w1_0, const float* __restrict__ w1_1,
    const float* __restrict__ w1_2,
    const float* __restrict__ w2_0, const float* __restrict__ w2_1,
    const float* __restrict__ w2_2,
    float* __restrict__ out, int num_tiles)
{
  // B fragments pre-packed in WMMA B layout: [mat(5)][kc(4)][lane(32)][h(16)]
  // mat 0..2 = w1_0 column blocks, mat 3 = w1_1, mat 4 = w1_2.
  // B layout: lane = column n (mod 16), lane group g holds K 8g..8g+7 then 16+8g..16+8g+7.
  __shared__ alignas(32) _Float16 ldsB[5 * 4 * 32 * 16];
  for (int e = threadIdx.x; e < 5 * 4 * 32 * 16; e += blockDim.x) {
    int h    = e & 15;
    int lane = (e >> 4) & 31;
    int kc   = (e >> 9) & 3;
    int mat  = e >> 11;
    int ncol = lane & 15;
    int k    = kc * 32 + (lane >> 4) * 8 + (h & 7) + (h >> 3) * 16;
    float v;
    if (mat < 3)       v = w1_0[k * 48 + mat * 16 + ncol];
    else if (mat == 3) v = w1_1[k * 16 + ncol];
    else               v = w1_2[k * 16 + ncol];
    ldsB[e] = (_Float16)v;
  }
  __syncthreads();

  const int lane = threadIdx.x & 31;
  const int wave = threadIdx.x >> 5;
  const int tile = blockIdx.x * (blockDim.x >> 5) + wave;
  if (tile >= num_tiles) return;   // wave-uniform: EXEC stays all-ones for WMMA

  const int node0 = tile * 16;
  const int m     = lane & 15;     // node row within tile / column index for B
  const int grp   = lane >> 4;
  const float* row = x + (size_t)(node0 + m) * D_IN;
  const v16h* Bf = (const v16h*)ldsB;

  v8f zero = {0.f, 0.f, 0.f, 0.f, 0.f, 0.f, 0.f, 0.f};
  v8f acc[11];
#pragma unroll
  for (int t = 0; t < 11; ++t) acc[t] = zero;

#pragma unroll
  for (int kc = 0; kc < 4; ++kc) {
    const int u1 = kc * 32 + grp * 8;
    v16h b0 = Bf[(0 * 4 + kc) * 32 + lane];
    v16h b1 = Bf[(1 * 4 + kc) * 32 + lane];
    v16h b2 = Bf[(2 * 4 + kc) * 32 + lane];
    v16h b3 = Bf[(3 * 4 + kc) * 32 + lane];
    v16h b4 = Bf[(4 * 4 + kc) * 32 + lane];

    // x0: one A fragment feeds three B tiles (y0 columns 0..47)
    {
      float buf[16];
      const v4f* p0 = (const v4f*)(row + u1);
      const v4f* p1 = (const v4f*)(row + u1 + 16);
#pragma unroll
      for (int t = 0; t < 2; ++t) {
        v4f v = __builtin_nontemporal_load(p0 + t);
        buf[4 * t + 0] = v[0]; buf[4 * t + 1] = v[1];
        buf[4 * t + 2] = v[2]; buf[4 * t + 3] = v[3];
      }
#pragma unroll
      for (int t = 0; t < 2; ++t) {
        v4f v = __builtin_nontemporal_load(p1 + t);
        buf[8 + 4 * t + 0] = v[0]; buf[8 + 4 * t + 1] = v[1];
        buf[8 + 4 * t + 2] = v[2]; buf[8 + 4 * t + 3] = v[3];
      }
      v16h a0;
#pragma unroll
      for (int h = 0; h < 16; ++h) a0[h] = (_Float16)buf[h];
      acc[0] = wmma_f16(a0, b0, acc[0]);
      acc[1] = wmma_f16(a0, b1, acc[1]);
      acc[2] = wmma_f16(a0, b2, acc[2]);
    }
    // x1: 3 interleaved components, range loaded exactly once
    run_group<3, 3>(row, 128, u1, b3, &acc[3]);
    // x2: 5 interleaved components, range loaded exactly once
    run_group<5, 5>(row, 512, u1, b4, &acc[6]);
  }

  // Second layer. C layout: acc[t][r] = Y_t[node r + grp*8, col m].
  const float w20 = w2_0[m];
  const float w21 = w2_1[m];
  const float w22 = w2_2[m];

#pragma unroll
  for (int r = 0; r < 8; ++r) {
    float s  = silu_f(acc[0][r] * INV_IN);   // silu(y0[:, :16])
    float g1 = silu_f(acc[1][r] * INV_IN);   // silu(y0[:, 16:32]) gates z1
    float g2 = silu_f(acc[2][r] * INV_IN);   // silu(y0[:, 32:48]) gates z2
    float p[9];
    p[0] = s * w20;
#pragma unroll
    for (int i = 0; i < 3; ++i) p[1 + i] = acc[3 + i][r] * INV_IN * g1 * w21;
#pragma unroll
    for (int i = 0; i < 5; ++i) p[4 + i] = acc[6 + i][r] * INV_IN * g2 * w22;
    // Sum over the 16 hidden columns: butterfly within each 16-lane half.
#pragma unroll
    for (int mask = 1; mask < 16; mask <<= 1) {
#pragma unroll
      for (int j = 0; j < 9; ++j) p[j] += __shfl_xor(p[j], mask, 32);
    }
    float v = p[0];
#pragma unroll
    for (int j = 1; j < 9; ++j) v = (m == j) ? p[j] : v;
    if (m < 9)
      out[(size_t)(node0 + r + grp * 8) * 9 + m] = v * INV_H;
  }
}

// Scalar fallback for nodes beyond the last full 16-node tile (unused for N=100000).
__global__ void readout_tail_kernel(
    const float* __restrict__ x,
    const float* __restrict__ w1_0, const float* __restrict__ w1_1,
    const float* __restrict__ w1_2,
    const float* __restrict__ w2_0, const float* __restrict__ w2_1,
    const float* __restrict__ w2_2,
    float* __restrict__ out, int start, int n)
{
  int node = start + blockIdx.x * blockDim.x + (int)threadIdx.x;
  if (node >= n) return;
  const float* row = x + (size_t)node * D_IN;
  float y0[48];
  for (int v = 0; v < 48; ++v) {
    float a = 0.f;
    for (int u = 0; u < 128; ++u) a += row[u] * w1_0[u * 48 + v];
    y0[v] = a * INV_IN;
  }
  float o[9];
  {
    float a = 0.f;
    for (int v = 0; v < 16; ++v) a += silu_f(y0[v]) * w2_0[v];
    o[0] = a * INV_H;
  }
  for (int i = 0; i < 3; ++i) {
    float a = 0.f;
    for (int v = 0; v < 16; ++v) {
      float y1 = 0.f;
      for (int u = 0; u < 128; ++u) y1 += row[128 + u * 3 + i] * w1_1[u * 16 + v];
      a += y1 * INV_IN * silu_f(y0[16 + v]) * w2_1[v];
    }
    o[1 + i] = a * INV_H;
  }
  for (int i = 0; i < 5; ++i) {
    float a = 0.f;
    for (int v = 0; v < 16; ++v) {
      float y2 = 0.f;
      for (int u = 0; u < 128; ++u) y2 += row[512 + u * 5 + i] * w1_2[u * 16 + v];
      a += y2 * INV_IN * silu_f(y0[32 + v]) * w2_2[v];
    }
    o[4 + i] = a * INV_H;
  }
  for (int j = 0; j < 9; ++j) out[(size_t)node * 9 + j] = o[j];
}

extern "C" void kernel_launch(void* const* d_in, const int* in_sizes, int n_in,
                              void* d_out, int out_size, void* d_ws, size_t ws_size,
                              hipStream_t stream) {
  (void)n_in; (void)out_size; (void)d_ws; (void)ws_size;
  const float* x    = (const float*)d_in[0];
  const float* w1_0 = (const float*)d_in[1];
  const float* w1_1 = (const float*)d_in[2];
  const float* w1_2 = (const float*)d_in[3];
  const float* w2_0 = (const float*)d_in[4];
  const float* w2_1 = (const float*)d_in[5];
  const float* w2_2 = (const float*)d_in[6];
  float* out = (float*)d_out;

  const int n = in_sizes[0] / D_IN;          // number of nodes
  const int tiles = n / 16;                  // full 16-node WMMA tiles
  if (tiles > 0) {
    const int wavesPerBlock = 8;             // 256 threads = 8 wave32
    const int blocks = (tiles + wavesPerBlock - 1) / wavesPerBlock;
    readout_wmma_kernel<<<blocks, wavesPerBlock * 32, 0, stream>>>(
        x, w1_0, w1_1, w1_2, w2_0, w2_1, w2_2, out, tiles);
  }
  const int rem = n - tiles * 16;
  if (rem > 0) {
    readout_tail_kernel<<<1, 64, 0, stream>>>(
        x, w1_0, w1_1, w1_2, w2_0, w2_1, w2_2, out, tiles * 16, n);
  }
}
